// GATPropagation_75445395521546
// MI455X (gfx1250) — matrix operand order repaired and checked by reading, not compiled, not run on hardware
//
#include <hip/hip_runtime.h>
#include <hip/hip_bf16.h>

typedef __attribute__((ext_vector_type(16))) _Float16 v16h;
typedef __attribute__((ext_vector_type(8)))  float    v8f;

#define N_NODES 20000
#define N_EDGES 320000
#define E_TOT   (N_EDGES + N_NODES)   // 340000 (with self-loops)
#define IN_CH   512
#define HEADS   4
#define OUT_CH  128
#define HC      (HEADS * OUT_CH)      // 512
#define KSTEPS  (IN_CH / 32)          // 16 k-steps of 32 for 16x16x32 WMMA
#define MTILES  (N_NODES / 16)        // 1250 (exact)
#define NTILES  (HC / 16)             // 32

// ---------------- monotonic f32 <-> u32 ordering for atomic segment-max ----
__device__ __forceinline__ unsigned enc_f32(float x) {
    unsigned u = __float_as_uint(x);
    return (u & 0x80000000u) ? ~u : (u | 0x80000000u);
}
__device__ __forceinline__ float dec_f32(unsigned k) {
    unsigned u = (k & 0x80000000u) ? (k & 0x7FFFFFFFu) : ~k;
    return __uint_as_float(u);
}

// ---------------- Pack x (f32 row-major) into WMMA A fragments (f16) -------
// A 16x32 f16 layout (ISA 7.12.2): lane L -> row m=L%16, half h=L/16;
// element e<8 : K = 32*ks + 8h + e ; e>=8 : K = 32*ks + 16 + 8h + (e-8)
__global__ void pack_x_f16(const float* __restrict__ x, _Float16* __restrict__ ap) {
    int t = blockIdx.x * blockDim.x + threadIdx.x;   // ((mt*KSTEPS+ks)*32+lane)
    if (t >= MTILES * KSTEPS * 32) return;
    int lane = t & 31;
    int ks   = (t >> 5) % KSTEPS;
    int mt   = (t >> 5) / KSTEPS;
    int m = lane & 15, h = lane >> 4;
    const float* row = x + (size_t)(mt * 16 + m) * IN_CH + ks * 32 + h * 8;
    v16h v;
#pragma unroll
    for (int e = 0; e < 8; ++e) v[e] = (_Float16)row[e];
#pragma unroll
    for (int e = 0; e < 8; ++e) v[8 + e] = (_Float16)row[16 + e];
    *(v16h*)(ap + (size_t)t * 16) = v;
}

// ---------------- Pack W (f32 [K=512][512] row-major) into B fragments -----
// B 32x16 f16 layout: lane L -> col n=L%16, half h=L/16; element e<8: K = 32*ks + 8h + e ; e>=8: K = 32*ks + 16 + 8h + (e-8)
__global__ void pack_w_f16(const float* __restrict__ W, _Float16* __restrict__ bp) {
    int t = blockIdx.x * blockDim.x + threadIdx.x;   // ((nt*KSTEPS+ks)*32+lane)
    if (t >= NTILES * KSTEPS * 32) return;
    int lane = t & 31;
    int ks   = (t >> 5) % KSTEPS;
    int nt   = (t >> 5) / KSTEPS;
    int n = lane & 15, h = lane >> 4;
    int col = nt * 16 + n;
    int k0  = ks * 32 + h * 8;
    v16h v;
#pragma unroll
    for (int e = 0; e < 16; ++e) v[e] = (_Float16)W[(size_t)(k0 + e + (e >= 8 ? 8 : 0)) * HC + col];
    *(v16h*)(bp + (size_t)t * 16) = v;
}

// ---------------- xl = x @ W via v_wmma_f32_16x16x32_f16 -------------------
// One block (8 wave32) per 16-row M-tile; wave w owns N-tiles [4w, 4w+4).
__global__ void __launch_bounds__(256) gemm_wmma(const _Float16* __restrict__ ap,
                                                 const _Float16* __restrict__ bp,
                                                 float* __restrict__ xl) {
    int mt   = blockIdx.x;
    int lane = threadIdx.x & 31;
    int wave = threadIdx.x >> 5;
    int nt0  = wave * 4;
    v8f c0 = {}, c1 = {}, c2 = {}, c3 = {};
    const _Float16* aB = ap + (size_t)mt * KSTEPS * 512 + lane * 16;
    const _Float16* bB = bp + lane * 16;
    for (int ks = 0; ks < KSTEPS; ++ks) {
        v16h a  = *(const v16h*)(aB + (size_t)ks * 512);
        v16h b0 = *(const v16h*)(bB + ((size_t)(nt0 + 0) * KSTEPS + ks) * 512);
        v16h b1 = *(const v16h*)(bB + ((size_t)(nt0 + 1) * KSTEPS + ks) * 512);
        v16h b2 = *(const v16h*)(bB + ((size_t)(nt0 + 2) * KSTEPS + ks) * 512);
        v16h b3 = *(const v16h*)(bB + ((size_t)(nt0 + 3) * KSTEPS + ks) * 512);
        c0 = __builtin_amdgcn_wmma_f32_16x16x32_f16(false, a, false, b0, (short)0, c0, false, false);
        c1 = __builtin_amdgcn_wmma_f32_16x16x32_f16(false, a, false, b1, (short)0, c1, false, false);
        c2 = __builtin_amdgcn_wmma_f32_16x16x32_f16(false, a, false, b2, (short)0, c2, false, false);
        c3 = __builtin_amdgcn_wmma_f32_16x16x32_f16(false, a, false, b3, (short)0, c3, false, false);
    }
    // C/D layout: element r -> row = r + 8*(lane/16), col = lane%16
    int n = lane & 15, h = lane >> 4;
#pragma unroll
    for (int r = 0; r < 8; ++r) {
        size_t row = (size_t)(mt * 16 + r + 8 * h) * HC;
        xl[row + (nt0 + 0) * 16 + n] = c0[r];
        xl[row + (nt0 + 1) * 16 + n] = c1[r];
        xl[row + (nt0 + 2) * 16 + n] = c2[r];
        xl[row + (nt0 + 3) * 16 + n] = c3[r];
    }
}

// ---------------- per-node attention halves a_s, a_d [N,H] -----------------
__global__ void __launch_bounds__(256) node_attn(const float* __restrict__ xl,
                                                 const float* __restrict__ att_src,
                                                 const float* __restrict__ att_dst,
                                                 float* __restrict__ a_s,
                                                 float* __restrict__ a_d) {
    int gw = blockIdx.x * 8 + (threadIdx.x >> 5);   // one wave per (node, head)
    int lane = threadIdx.x & 31;
    if (gw >= N_NODES * HEADS) return;
    int node = gw >> 2, h = gw & 3;
    const float* xr = xl + (size_t)node * HC + h * OUT_CH;
    const float* as = att_src + h * OUT_CH;
    const float* ad = att_dst + h * OUT_CH;
    float s = 0.f, d = 0.f;
#pragma unroll
    for (int i = 0; i < 4; ++i) {
        int c = lane + 32 * i;
        float v = xr[c];
        s += v * as[c];
        d += v * ad[c];
    }
#pragma unroll
    for (int off = 16; off > 0; off >>= 1) {
        s += __shfl_xor(s, off, 32);
        d += __shfl_xor(d, off, 32);
    }
    if (lane == 0) { a_s[gw] = s; a_d[gw] = d; }
}

// ---------------- edge logits + atomic segment max over dst ----------------
__global__ void edge_logits(const long long* __restrict__ ei,
                            const float* __restrict__ a_s, const float* __restrict__ a_d,
                            float* __restrict__ alpha, unsigned* __restrict__ amax) {
    int t = blockIdx.x * blockDim.x + threadIdx.x;
    if (t >= E_TOT * HEADS) return;
    int e = t >> 2, h = t & 3;
    int s, d;
    if (e < N_EDGES) { s = (int)ei[e]; d = (int)ei[N_EDGES + e]; }
    else             { s = d = e - N_EDGES; }            // self-loop
    float v = a_s[s * HEADS + h] + a_d[d * HEADS + h];
    v = v > 0.f ? v : 0.2f * v;                          // leaky_relu(0.2)
    alpha[t] = v;
    atomicMax(&amax[d * HEADS + h], enc_f32(v));
}

// ---------------- exp(alpha - max) + atomic segment sum --------------------
__global__ void edge_exp(const long long* __restrict__ ei,
                         const unsigned* __restrict__ amax,
                         float* __restrict__ alpha, float* __restrict__ denom) {
    int t = blockIdx.x * blockDim.x + threadIdx.x;
    if (t >= E_TOT * HEADS) return;
    int e = t >> 2, h = t & 3;
    int d = (e < N_EDGES) ? (int)ei[N_EDGES + e] : (e - N_EDGES);
    float m  = dec_f32(amax[d * HEADS + h]);
    float ae = expf(alpha[t] - m);
    alpha[t] = ae;
    atomicAdd(&denom[d * HEADS + h], ae);
}

// ---------------- weighted scatter out[dst] += alpha * xl[src] -------------
// One wave per (edge, head); lanes cover the 128 channels (coalesced).
__global__ void __launch_bounds__(256) edge_scatter(const long long* __restrict__ ei,
                                                    const float* __restrict__ xl,
                                                    const float* __restrict__ alpha,
                                                    const float* __restrict__ denom,
                                                    float* __restrict__ acc) {
    int gw = blockIdx.x * 8 + (threadIdx.x >> 5);
    int lane = threadIdx.x & 31;
    if (gw >= E_TOT * HEADS) return;
    int e = gw >> 2, h = gw & 3;
    int s, d;
    if (e < N_EDGES) { s = (int)ei[e]; d = (int)ei[N_EDGES + e]; }
    else             { s = d = e - N_EDGES; }
    float w = alpha[gw] / (denom[d * HEADS + h] + 1e-16f);
    const float* xs = xl  + (size_t)s * HC + h * OUT_CH;
    float*       ao = acc + (size_t)d * HC + h * OUT_CH;
#pragma unroll
    for (int i = 0; i < 4; ++i) {
        int c = lane + 32 * i;
        atomicAdd(&ao[c], w * xs[c]);
    }
}

// ---------------- head mean + bias + ELU -----------------------------------
__global__ void finalize(const float* __restrict__ acc, const float* __restrict__ bias,
                         float* __restrict__ out) {
    int t = blockIdx.x * blockDim.x + threadIdx.x;
    if (t >= N_NODES * OUT_CH) return;
    int n = t >> 7, c = t & 127;
    const float* a = acc + (size_t)n * HC;
    float v = 0.25f * (a[c] + a[c + 128] + a[c + 256] + a[c + 384]) + bias[c];
    out[t] = v > 0.f ? v : (expf(v) - 1.0f);
}

extern "C" void kernel_launch(void* const* d_in, const int* in_sizes, int n_in,
                              void* d_out, int out_size, void* d_ws, size_t ws_size,
                              hipStream_t stream) {
    const float*     x    = (const float*)d_in[0];
    const float*     W    = (const float*)d_in[1];
    const float*     atts = (const float*)d_in[2];
    const float*     attd = (const float*)d_in[3];
    const float*     bias = (const float*)d_in[4];
    const long long* ei   = (const long long*)d_in[5];

    char* ws = (char*)d_ws;
    _Float16* Apack = (_Float16*)ws;  ws += (size_t)N_NODES * IN_CH * 2;      // 20.48 MB
    _Float16* Bpack = (_Float16*)ws;  ws += (size_t)IN_CH * HC * 2;           //  0.52 MB
    float*    xl    = (float*)ws;     ws += (size_t)N_NODES * HC * 4;         // 40.96 MB
    float*    a_s   = (float*)ws;     ws += (size_t)N_NODES * HEADS * 4;
    float*    a_d   = (float*)ws;     ws += (size_t)N_NODES * HEADS * 4;
    float*    alpha = (float*)ws;     ws += (size_t)E_TOT * HEADS * 4;        //  5.44 MB
    unsigned* amax  = (unsigned*)ws;  ws += (size_t)N_NODES * HEADS * 4;      // | contiguous
    float*    denom = (float*)ws;     ws += (size_t)N_NODES * HEADS * 4;      // | zero block
    float*    acc   = (float*)ws;     ws += (size_t)N_NODES * HC * 4;         // | 41.6 MB

    // zero amax + denom + acc in one capturable memset (contiguous region)
    size_t zbytes = (size_t)N_NODES * HEADS * 4 * 2 + (size_t)N_NODES * HC * 4;
    hipMemsetAsync(amax, 0, zbytes, stream);

    pack_x_f16 <<<(MTILES * KSTEPS * 32 + 255) / 256, 256, 0, stream>>>(x, Apack);
    pack_w_f16 <<<(NTILES * KSTEPS * 32 + 255) / 256, 256, 0, stream>>>(W, Bpack);
    gemm_wmma  <<<MTILES, 256, 0, stream>>>(Apack, Bpack, xl);
    node_attn  <<<(N_NODES * HEADS) / 8, 256, 0, stream>>>(xl, atts, attd, a_s, a_d);
    edge_logits<<<(E_TOT * HEADS + 255) / 256, 256, 0, stream>>>(ei, a_s, a_d, alpha, amax);
    edge_exp   <<<(E_TOT * HEADS + 255) / 256, 256, 0, stream>>>(ei, amax, alpha, denom);
    edge_scatter<<<(E_TOT * HEADS) / 8, 256, 0, stream>>>(ei, xl, alpha, denom, acc);
    finalize   <<<(N_NODES * OUT_CH + 255) / 256, 256, 0, stream>>>(acc, bias, (float*)d_out);
}
